// AttnBlock_46428596470479
// MI455X (gfx1250) — matrix operand order repaired and checked by reading, not compiled, not run on hardware
//
#include <hip/hip_runtime.h>
#include <hip/hip_bf16.h>

// ---- shapes (fixed by the reference) ----
#define B_  16
#define C_  512
#define N_  1024   // H*W
#define G_  32     // groups

#define AROW 40    // padded LDS row length in bf16 elements (80B, 16B-aligned)

typedef __attribute__((ext_vector_type(16))) __bf16 v16bf;
typedef __attribute__((ext_vector_type(8)))  __bf16 v8bf;
typedef __attribute__((ext_vector_type(8)))  float  v8f;

// ---------------------------------------------------------------------------
// WMMA helpers
// ---------------------------------------------------------------------------
static __device__ __forceinline__
v8f wmma_bf16(v16bf a, v16bf b, v8f c) {
  // (neg_a, A, neg_b, B, c_mod, C, reuse_a, reuse_b)
  return __builtin_amdgcn_wmma_f32_16x16x32_bf16(false, a, false, b,
                                                 (short)0, c, false, false);
}

// A-fragment from LDS tile (rows of 32 K-elements, padded to AROW):
// lane L holds row M = L&15; lanes 0-15 K={0..7,16..23}, lanes 16-31 K={8..15,24..31}.
static __device__ __forceinline__
v16bf lds_a_frag(const __bf16* __restrict__ As, int m0w) {
  int lane = threadIdx.x & 31;
  int row  = m0w + (lane & 15);
  int kb   = (lane >> 4) << 3;
  const __bf16* p = As + row * AROW + kb;
  v8bf lo = *(const v8bf*)(p);
  v8bf hi = *(const v8bf*)(p + 16);
  v16bf r;
#pragma unroll
  for (int i = 0; i < 8; ++i) { r[i] = lo[i]; r[i + 8] = hi[i]; }
  return r;
}

// B-fragment from LDS tile holding B^T rows (N rows x 32 K, padded to AROW):
// lane L holds column N = L&15; lanes 0-15 K={0..7,16..23}, lanes 16-31 K={8..15,24..31}.
static __device__ __forceinline__
v16bf lds_b_frag(const __bf16* __restrict__ Bs, int n0w) {
  int lane = threadIdx.x & 31;
  int col  = n0w + (lane & 15);
  int kb   = (lane >> 4) << 3;
  const __bf16* p = Bs + col * AROW + kb;
  v8bf lo = *(const v8bf*)(p);
  v8bf hi = *(const v8bf*)(p + 16);
  v16bf r;
#pragma unroll
  for (int i = 0; i < 8; ++i) { r[i] = lo[i]; r[i + 8] = hi[i]; }
  return r;
}

// ---------------------------------------------------------------------------
// Async global -> LDS 16-byte copy (CDNA5 VGLOBAL async path, ASYNCcnt).
// GVS mode: 64-bit SGPR base + 32-bit VGPR byte offset; LDS dest addr in VGPR.
// ---------------------------------------------------------------------------
static __device__ __forceinline__
void async_copy16(unsigned lds_addr, unsigned voff, unsigned long long sbase) {
  asm volatile("global_load_async_to_lds_b128 %0, %1, %2"
               :: "v"(lds_addr), "v"(voff), "s"(sbase)
               : "memory");
}

static __device__ __forceinline__ void wait_async0() {
  asm volatile("s_wait_asynccnt 0x0" ::: "memory");
}

static __device__ __forceinline__ unsigned lds_addr_of(const void* p) {
  // generic pointer to LDS: low 32 bits are the LDS byte offset
  return (unsigned)(size_t)p;
}

// One wave computes a 32x32 output tile (2x2 grid of 16x16 WMMA tiles).
struct Acc4 { v8f c00, c01, c10, c11; };

// ---------------------------------------------------------------------------
// Workgroup-cooperative GEMM: 256 threads (8 waves, 4x2), macro tile 128x64.
// A (MxK) row-major; B given as B^T (NxK) row-major. Double-buffered LDS
// staging via async copies; each wave returns its 32x32 accumulator block.
// Wave tile origin: (m0 + (wave>>1)*32, n0 + (wave&1)*32).
// ---------------------------------------------------------------------------
static __device__ __forceinline__
Acc4 gemm_block(const __bf16* __restrict__ A, int lda,
                const __bf16* __restrict__ Bt, int ldb,
                int m0, int n0, int K) {
  __shared__ __bf16 Asm[2][128 * AROW];  // 2 x 10KB
  __shared__ __bf16 Bsm[2][64 * AROW];   // 2 x  5KB

  int t = threadIdx.x;

  // A tile: 128 rows x 32 cols bf16 = 512 16B-chunks; thread t owns chunks t, t+256.
  int ar0 = t >> 2;                 // rows 0..63
  int ar1 = (t + 256) >> 2;         // rows 64..127
  int ae  = (t & 3) * 8;            // element offset within row (0,8,16,24)
  // B tile: 64 rows x 32 cols = 256 chunks; thread t owns chunk t.
  int br = t >> 2;
  int be = (t & 3) * 8;

  unsigned long long aBase = (unsigned long long)(const void*)(A + (size_t)m0 * lda);
  unsigned long long bBase = (unsigned long long)(const void*)(Bt + (size_t)n0 * ldb);
  unsigned aOff0 = (unsigned)((ar0 * lda + ae) * 2);
  unsigned aOff1 = (unsigned)((ar1 * lda + ae) * 2);
  unsigned bOff  = (unsigned)((br  * ldb + be) * 2);

  unsigned aLds0[2], aLds1[2], bLds[2];
#pragma unroll
  for (int bb = 0; bb < 2; ++bb) {
    aLds0[bb] = lds_addr_of(&Asm[bb][ar0 * AROW + ae]);
    aLds1[bb] = lds_addr_of(&Asm[bb][ar1 * AROW + ae]);
    bLds[bb]  = lds_addr_of(&Bsm[bb][br  * AROW + be]);
  }

  int wave = t >> 5;
  int m0w  = (wave >> 1) * 32;
  int n0w  = (wave & 1) * 32;

  v8f z = {0.f, 0.f, 0.f, 0.f, 0.f, 0.f, 0.f, 0.f};
  Acc4 acc; acc.c00 = z; acc.c01 = z; acc.c10 = z; acc.c11 = z;

  // prologue: stage k=0 into buffer 0
  async_copy16(aLds0[0], aOff0, aBase);
  async_copy16(aLds1[0], aOff1, aBase);
  async_copy16(bLds[0],  bOff,  bBase);
  wait_async0();
  __syncthreads();

  int cur = 0;
  for (int k = 0; k < K; k += 32) {
    int nxt = cur ^ 1;
    if (k + 32 < K) {
      unsigned kk = (unsigned)((k + 32) * 2);
      async_copy16(aLds0[nxt], aOff0 + kk, aBase);
      async_copy16(aLds1[nxt], aOff1 + kk, aBase);
      async_copy16(bLds[nxt],  bOff  + kk, bBase);
    }
    const __bf16* As = Asm[cur];
    const __bf16* Bs = Bsm[cur];
    v16bf a0 = lds_a_frag(As, m0w);
    v16bf a1 = lds_a_frag(As, m0w + 16);
    v16bf b0 = lds_b_frag(Bs, n0w);
    v16bf b1 = lds_b_frag(Bs, n0w + 16);
    acc.c00 = wmma_bf16(a0, b0, acc.c00);
    acc.c01 = wmma_bf16(a0, b1, acc.c01);
    acc.c10 = wmma_bf16(a1, b0, acc.c10);
    acc.c11 = wmma_bf16(a1, b1, acc.c11);
    if (k + 32 < K) wait_async0();   // this wave's copies into nxt complete
    __syncthreads();                 // all waves done reading cur / copying nxt
    cur = nxt;
  }
  return acc;
}

// ---------------------------------------------------------------------------
// Kernel 0: convert weights f32 -> bf16
// ---------------------------------------------------------------------------
__global__ void k_cvt_weights(const float* __restrict__ qkv_w,
                              const float* __restrict__ proj_w,
                              __bf16* __restrict__ wq,
                              __bf16* __restrict__ wp) {
  int i = blockIdx.x * 256 + threadIdx.x;
  if (i < 3 * C_ * C_) wq[i] = (__bf16)qkv_w[i];
  if (i < C_ * C_)     wp[i] = (__bf16)proj_w[i];
}

// ---------------------------------------------------------------------------
// Kernel 1: GroupNorm; writes h transposed (per batch: N_ x C_, bf16).
// One block per (batch, group): 16 channels x 1024 spatial.
// ---------------------------------------------------------------------------
__global__ void k_groupnorm(const float* __restrict__ x,
                            const float* __restrict__ nw,
                            const float* __restrict__ nb,
                            __bf16* __restrict__ ht) {
  __shared__ float red[256];
  int b = blockIdx.x >> 5;
  int g = blockIdx.x & 31;
  const float* xp = x + ((size_t)b * C_ + g * 16) * N_;

  float s = 0.f, s2 = 0.f;
  for (int i = threadIdx.x; i < 16 * N_; i += 256) {
    float v = xp[i]; s += v; s2 += v * v;
  }
  red[threadIdx.x] = s; __syncthreads();
  for (int off = 128; off > 0; off >>= 1) {
    if ((int)threadIdx.x < off) red[threadIdx.x] += red[threadIdx.x + off];
    __syncthreads();
  }
  float S = red[0]; __syncthreads();
  red[threadIdx.x] = s2; __syncthreads();
  for (int off = 128; off > 0; off >>= 1) {
    if ((int)threadIdx.x < off) red[threadIdx.x] += red[threadIdx.x + off];
    __syncthreads();
  }
  float S2 = red[0];

  float mu   = S  * (1.f / 16384.f);
  float var  = S2 * (1.f / 16384.f) - mu * mu;
  float rstd = rsqrtf(var + 1e-6f);

  for (int i = threadIdx.x; i < 16 * N_; i += 256) {
    int cl = i >> 10;
    int n  = i & (N_ - 1);
    int c  = g * 16 + cl;
    float v = (xp[i] - mu) * rstd * nw[c] + nb[c];
    ht[((size_t)b * N_ + n) * C_ + c] = (__bf16)v;
  }
}

// ---------------------------------------------------------------------------
// Kernel 2: QKV GEMM. A = wq (1536x512), B^T = h^T (1024x512).
// Epilogue scatters into Q^T (scaled), K^T (NxC) and V (CxN), bf16.
// Grid: 16 batches * (1536/128)*(1024/64) = 3072 blocks.
// ---------------------------------------------------------------------------
__global__ void k_qkv(const __bf16* __restrict__ wq,
                      const __bf16* __restrict__ ht,
                      const float* __restrict__ qkv_b,
                      __bf16* __restrict__ qt,
                      __bf16* __restrict__ kt,
                      __bf16* __restrict__ vv) {
  int b  = blockIdx.x / 192;
  int r  = blockIdx.x % 192;
  int m0 = (r >> 4) * 128;
  int n0 = (r & 15) * 64;

  const __bf16* Bt = ht + (size_t)b * N_ * C_;
  Acc4 acc = gemm_block(wq, C_, Bt, C_, m0, n0, C_);

  int wave = threadIdx.x >> 5;
  int mW = m0 + (wave >> 1) * 32;
  int nW = n0 + (wave & 1) * 32;
  int lane = threadIdx.x & 31;
  int mb = (lane >> 4) << 3;
  int nn = lane & 15;
  const float scale = 0.04419417382415922f;   // 512^-0.5 folded into Q

  auto emit = [&](const v8f& c, int mt, int nt) {
#pragma unroll
    for (int i = 0; i < 8; ++i) {
      int o = mt + mb + i;
      int n = nt + nn;
      float val = c[i] + qkv_b[o];
      size_t bn = (size_t)b * N_ + n;
      if (o < C_)            qt[bn * C_ + o]        = (__bf16)(val * scale);
      else if (o < 2 * C_)   kt[bn * C_ + (o - C_)] = (__bf16)val;
      else                   vv[((size_t)b * C_ + (o - 2 * C_)) * N_ + n] = (__bf16)val;
    }
  };
  emit(acc.c00, mW, nW);       emit(acc.c01, mW, nW + 16);
  emit(acc.c10, mW + 16, nW);  emit(acc.c11, mW + 16, nW + 16);
}

// ---------------------------------------------------------------------------
// Kernel 3: S[b,i,j] = sum_c Qs[c,i] K[c,j]. A = Q^T, B^T = K^T, out f32.
// Grid: 16 * (1024/128)*(1024/64) = 2048 blocks.
// ---------------------------------------------------------------------------
__global__ void k_scores(const __bf16* __restrict__ qt,
                         const __bf16* __restrict__ kt,
                         float* __restrict__ S) {
  int b  = blockIdx.x / 128;
  int r  = blockIdx.x % 128;
  int m0 = (r >> 4) * 128;
  int n0 = (r & 15) * 64;

  const __bf16* A  = qt + (size_t)b * N_ * C_;
  const __bf16* Bt = kt + (size_t)b * N_ * C_;
  Acc4 acc = gemm_block(A, C_, Bt, C_, m0, n0, C_);

  int wave = threadIdx.x >> 5;
  int mW = m0 + (wave >> 1) * 32;
  int nW = n0 + (wave & 1) * 32;
  int lane = threadIdx.x & 31;
  int mb = (lane >> 4) << 3;
  int nn = lane & 15;
  float* Sp = S + (size_t)b * N_ * N_;
  auto emit = [&](const v8f& c, int mt, int nt) {
#pragma unroll
    for (int i = 0; i < 8; ++i)
      Sp[(size_t)(mt + mb + i) * N_ + (nt + nn)] = c[i];
  };
  emit(acc.c00, mW, nW);       emit(acc.c01, mW, nW + 16);
  emit(acc.c10, mW + 16, nW);  emit(acc.c11, mW + 16, nW + 16);
}

// ---------------------------------------------------------------------------
// Kernel 4: row softmax over j; P (bf16) = softmax(S). One block per row.
// ---------------------------------------------------------------------------
__global__ void k_softmax(const float* __restrict__ S, __bf16* __restrict__ P) {
  __shared__ float red[256];
  size_t row = blockIdx.x;
  const float* sp = S + row * N_;
  __bf16* pp = P + row * N_;

  float v[4];
  float m = -1e30f;
#pragma unroll
  for (int t = 0; t < 4; ++t) { v[t] = sp[threadIdx.x + t * 256]; m = fmaxf(m, v[t]); }
  red[threadIdx.x] = m; __syncthreads();
  for (int off = 128; off > 0; off >>= 1) {
    if ((int)threadIdx.x < off)
      red[threadIdx.x] = fmaxf(red[threadIdx.x], red[threadIdx.x + off]);
    __syncthreads();
  }
  m = red[0]; __syncthreads();

  float s = 0.f;
#pragma unroll
  for (int t = 0; t < 4; ++t) { v[t] = __expf(v[t] - m); s += v[t]; }
  red[threadIdx.x] = s; __syncthreads();
  for (int off = 128; off > 0; off >>= 1) {
    if ((int)threadIdx.x < off) red[threadIdx.x] += red[threadIdx.x + off];
    __syncthreads();
  }
  float inv = 1.f / red[0];
#pragma unroll
  for (int t = 0; t < 4; ++t)
    pp[threadIdx.x + t * 256] = (__bf16)(v[t] * inv);
}

// ---------------------------------------------------------------------------
// Kernel 5: O[b,c,i] = sum_j V[c,j] P[i,j]. A = V (512x1024), B^T = P.
// Stores O transposed (N_ x C_) for the proj GEMM. Grid: 16*4*16 = 1024.
// ---------------------------------------------------------------------------
__global__ void k_attn_o(const __bf16* __restrict__ vv,
                         const __bf16* __restrict__ P,
                         __bf16* __restrict__ ot) {
  int b  = blockIdx.x / 64;
  int r  = blockIdx.x % 64;
  int m0 = (r >> 4) * 128;
  int n0 = (r & 15) * 64;

  const __bf16* A  = vv + (size_t)b * C_ * N_;
  const __bf16* Bt = P  + (size_t)b * N_ * N_;
  Acc4 acc = gemm_block(A, N_, Bt, N_, m0, n0, N_);

  int wave = threadIdx.x >> 5;
  int mW = m0 + (wave >> 1) * 32;
  int nW = n0 + (wave & 1) * 32;
  int lane = threadIdx.x & 31;
  int mb = (lane >> 4) << 3;
  int nn = lane & 15;
  __bf16* op = ot + (size_t)b * N_ * C_;
  auto emit = [&](const v8f& c, int mt, int nt) {
#pragma unroll
    for (int i = 0; i < 8; ++i)
      op[(size_t)(nt + nn) * C_ + (mt + mb + i)] = (__bf16)c[i];
  };
  emit(acc.c00, mW, nW);       emit(acc.c01, mW, nW + 16);
  emit(acc.c10, mW + 16, nW);  emit(acc.c11, mW + 16, nW + 16);
}

// ---------------------------------------------------------------------------
// Kernel 6: out = x + proj_w @ O + proj_b. A = wp (512x512), B^T = Ot.
// Grid: 16*4*16 = 1024.
// ---------------------------------------------------------------------------
__global__ void k_proj(const __bf16* __restrict__ wp,
                       const __bf16* __restrict__ ot,
                       const float* __restrict__ proj_b,
                       const float* __restrict__ x,
                       float* __restrict__ out) {
  int b  = blockIdx.x / 64;
  int r  = blockIdx.x % 64;
  int m0 = (r >> 4) * 128;
  int n0 = (r & 15) * 64;

  const __bf16* Bt = ot + (size_t)b * N_ * C_;
  Acc4 acc = gemm_block(wp, C_, Bt, C_, m0, n0, C_);

  int wave = threadIdx.x >> 5;
  int mW = m0 + (wave >> 1) * 32;
  int nW = n0 + (wave & 1) * 32;
  int lane = threadIdx.x & 31;
  int mb = (lane >> 4) << 3;
  int nn = lane & 15;
  auto emit = [&](const v8f& c, int mt, int nt) {
#pragma unroll
    for (int i = 0; i < 8; ++i) {
      int o = mt + mb + i;
      int n = nt + nn;
      size_t idx = ((size_t)b * C_ + o) * N_ + n;
      out[idx] = x[idx] + c[i] + proj_b[o];
    }
  };
  emit(acc.c00, mW, nW);       emit(acc.c01, mW, nW + 16);
  emit(acc.c10, mW + 16, nW);  emit(acc.c11, mW + 16, nW + 16);
}

// ---------------------------------------------------------------------------
extern "C" void kernel_launch(void* const* d_in, const int* in_sizes, int n_in,
                              void* d_out, int out_size, void* d_ws, size_t ws_size,
                              hipStream_t stream) {
  (void)in_sizes; (void)n_in; (void)out_size; (void)ws_size;
  const float* x      = (const float*)d_in[0];
  const float* norm_w = (const float*)d_in[1];
  const float* norm_b = (const float*)d_in[2];
  const float* qkv_w  = (const float*)d_in[3];
  const float* qkv_b  = (const float*)d_in[4];
  const float* proj_w = (const float*)d_in[5];
  const float* proj_b = (const float*)d_in[6];
  float* out = (float*)d_out;

  char* ws = (char*)d_ws;
  size_t off = 0;
  auto alloc = [&](size_t bytes) -> void* {
    void* p = ws + off;
    off += (bytes + 255) & ~(size_t)255;
    return p;
  };
  __bf16* wq = (__bf16*)alloc((size_t)3 * C_ * C_ * 2);  //  1.5 MB
  __bf16* wp = (__bf16*)alloc((size_t)C_ * C_ * 2);      //  0.5 MB
  __bf16* ht = (__bf16*)alloc((size_t)B_ * N_ * C_ * 2); // 16.8 MB
  __bf16* qt = (__bf16*)alloc((size_t)B_ * N_ * C_ * 2); // 16.8 MB
  __bf16* kt = (__bf16*)alloc((size_t)B_ * N_ * C_ * 2); // 16.8 MB
  __bf16* vv = (__bf16*)alloc((size_t)B_ * C_ * N_ * 2); // 16.8 MB
  float*  S  = (float*) alloc((size_t)B_ * N_ * N_ * 4); // 67.1 MB
  __bf16* P  = (__bf16*)alloc((size_t)B_ * N_ * N_ * 2); // 33.6 MB
  __bf16* ot = ht;   // h^T dead after k_qkv; reuse for O^T

  k_cvt_weights<<<3072, 256, 0, stream>>>(qkv_w, proj_w, wq, wp);
  k_groupnorm  <<<B_ * G_, 256, 0, stream>>>(x, norm_w, norm_b, ht);
  k_qkv        <<<3072, 256, 0, stream>>>(wq, ht, qkv_b, qt, kt, vv);
  k_scores     <<<2048, 256, 0, stream>>>(qt, kt, S);
  k_softmax    <<<B_ * N_, 256, 0, stream>>>(S, P);
  k_attn_o     <<<1024, 256, 0, stream>>>(vv, P, ot);
  k_proj       <<<1024, 256, 0, stream>>>(wp, ot, proj_b, x, out);
}
